// _BiMamba2Block_7919919694489
// MI455X (gfx1250) — matrix operand order repaired and checked by reading, not compiled, not run on hardware
//
#include <hip/hip_runtime.h>
#include <hip/hip_bf16.h>

// ---------------- problem constants ----------------
#define BATCH 16
#define SEQ   2048
#define DM    128   // D_MODEL
#define DI    256   // D_INNER
#define DS    64    // D_STATE
#define NH    4     // NHEADS
#define HD    64    // HEADDIM
#define CH    64    // CHUNK
#define NC    32    // SEQ/CHUNK
#define NPJ   644   // D_IN_PROJ
#define NPP   704   // padded to 11*64 for WMMA tiling

typedef __bf16 bf16_t;
typedef __attribute__((ext_vector_type(16))) __bf16 v16bf;
typedef __attribute__((ext_vector_type(8)))  float  v8f;

// ---------------- bf16 helpers (RNE) ----------------
__device__ inline bf16_t f2bf(float f) {
  union { float f; unsigned u; } v; v.f = f;
  unsigned r = v.u + 0x7fffu + ((v.u >> 16) & 1u);
  union { unsigned short s; bf16_t b; } o; o.s = (unsigned short)(r >> 16);
  return o.b;
}
__device__ inline float bf2f(bf16_t b) {
  union { unsigned short s; bf16_t b; } i; i.b = b;
  union { unsigned u; float f; } o; o.u = ((unsigned)i.s) << 16;
  return o.f;
}

// ---------------- WMMA fragment helpers (CDNA5 16x16x32 bf16 layouts) ----------
// A (16x32, MxK): lanes 0-15 => M=lane, elems 0..7 = K 0..7, elems 8..15 = K 16..23
//                 lanes 16-31 => M=lane-16, elems 0..7 = K 8..15, elems 8..15 = K 24..31
__device__ inline v16bf load_a_frag(const bf16_t* src, int rs, int cs, int lane) {
  const int m  = lane & 15;
  const int k0 = (lane < 16) ? 0 : 8;
  v16bf a;
#pragma unroll
  for (int e = 0; e < 8; ++e) {
    a[e]     = src[m * rs + (k0 + e) * cs];
    a[e + 8] = src[m * rs + (k0 + 16 + e) * cs];
  }
  return a;
}
// B (32x16, KxN): lanes 0-15 => N=lane, elems 0..7 = K 0..7, elems 8..15 = K 16..23; lanes 16-31 => K 8..15, 24..31
__device__ inline v16bf load_b_frag(const bf16_t* src, int rs, int cs, int lane) {
  const int n  = lane & 15;
  const int kb = (lane < 16) ? 0 : 8;
  v16bf b;
#pragma unroll
  for (int e = 0; e < 8; ++e) {
    b[e]     = src[(kb + e) * rs + n * cs];
    b[e + 8] = src[(kb + 16 + e) * rs + n * cs];
  }
  return b;
}
// C/D (16x16 f32): lanes 0-15 => N=lane, VGPR v => M=v; lanes 16-31 => M=8+v
__device__ inline void store_c_f32(float* dst, int ldc, int lane, v8f c) {
  const int n  = lane & 15;
  const int mb = (lane < 16) ? 0 : 8;
#pragma unroll
  for (int v = 0; v < 8; ++v) dst[(mb + v) * ldc + n] = c[v];
}
__device__ inline void accum_c_f32(float* dst, int ldc, int lane, v8f c) {
  const int n  = lane & 15;
  const int mb = (lane < 16) ? 0 : 8;
#pragma unroll
  for (int v = 0; v < 8; ++v) dst[(mb + v) * ldc + n] += c[v];
}
__device__ inline void store_c_bf(bf16_t* dst, int ldc, int lane, v8f c) {
  const int n  = lane & 15;
  const int mb = (lane < 16) ? 0 : 8;
#pragma unroll
  for (int v = 0; v < 8; ++v) dst[(mb + v) * ldc + n] = f2bf(c[v]);
}
__device__ inline v8f wmma_bf16(v16bf a, v16bf b, v8f c) {
  return __builtin_amdgcn_wmma_f32_16x16x32_bf16(false, a, false, b, (short)0, c,
                                                 false, false);
}

// ---------------- K0: convert x to bf16, forward + flipped copies ----------------
__global__ void kprep_x(const float* __restrict__ x, bf16_t* __restrict__ xbf) {
  int idx = blockIdx.x * blockDim.x + threadIdx.x;
  if (idx >= 2 * BATCH * SEQ * DM) return;
  int m = idx % DM; int t = idx / DM;
  int l = t % SEQ;  t /= SEQ;
  int b = t % BATCH; int d = t / BATCH;
  int ls = d ? (SEQ - 1 - l) : l;
  xbf[idx] = f2bf(x[(b * SEQ + ls) * DM + m]);
}
__global__ void kprep_wi(const float* __restrict__ wf, const float* __restrict__ wb,
                         bf16_t* __restrict__ wib) {
  int idx = blockIdx.x * blockDim.x + threadIdx.x;
  if (idx >= 2 * NPP * DM) return;
  int k = idx % DM; int t = idx / DM;
  int n = t % NPP;  int d = t / NPP;
  const float* w = d ? wb : wf;
  wib[idx] = f2bf((n < NPJ) ? w[n * DM + k] : 0.0f);
}
__global__ void kprep_wo(const float* __restrict__ wf, const float* __restrict__ wb,
                         bf16_t* __restrict__ wob) {
  int idx = blockIdx.x * blockDim.x + threadIdx.x;
  if (idx >= 2 * DM * DI) return;
  int k = idx % DI; int t = idx / DI;
  int n = t % DM;   int d = t / DM;
  wob[idx] = f2bf((d ? wb : wf)[n * DI + k]);
}

// ---------------- K1: in-proj GEMM  zxb[row, 0:704] = xbf[row,:] @ WiB^T -------
// block = 4 waves; wave w = rows (blockIdx.x*64 + w*16 .. +16) x 64 cols, K=128
__global__ void kgemm_in(const bf16_t* __restrict__ xbf, const bf16_t* __restrict__ wib,
                         bf16_t* __restrict__ zxb) {
  const int lane = threadIdx.x & 31;
  const int w    = threadIdx.x >> 5;
  const int row0 = blockIdx.x * 64 + w * 16;
  const int col0 = blockIdx.y * 64;
  const int d = row0 / (BATCH * SEQ);
  const bf16_t* A  = xbf + (size_t)row0 * DM;
  const bf16_t* Wd = wib + (size_t)d * NPP * DM;
  v8f acc[4] = {};
  for (int ks = 0; ks < DM; ks += 32) {
    __builtin_prefetch(A + ks + 32, 0, 1);
    v16bf a = load_a_frag(A + ks, DM, 1, lane);
#pragma unroll
    for (int j = 0; j < 4; ++j) {
      v16bf bb = load_b_frag(Wd + (size_t)(col0 + 16 * j) * DM + ks, 1, DM, lane);
      acc[j] = wmma_bf16(a, bb, acc[j]);
    }
  }
#pragma unroll
  for (int j = 0; j < 4; ++j)
    store_c_bf(zxb + (size_t)row0 * NPP + col0 + 16 * j, NPP, lane, acc[j]);
}

// ---------------- K2: dt softplus, conv1d+SiLU, splits, Y init = D*xs ----------
__global__ void kpost(const bf16_t* __restrict__ zxb,
                      const float* cwf, const float* cbf, const float* dtbf,
                      const float* Alf, const float* Dpf,
                      const float* cwb, const float* cbb, const float* dtbb,
                      const float* Alb, const float* Dpb,
                      bf16_t* __restrict__ xdt, bf16_t* __restrict__ Bb,
                      bf16_t* __restrict__ Cb, float* __restrict__ dAG,
                      float* __restrict__ Yf) {
  __shared__ float dt_s[NH];
  const int row = blockIdx.x;          // d*BATCH*SEQ flattened
  const int l   = row % SEQ;
  const int db  = row / SEQ;           // d*BATCH + b
  const int d   = db / BATCH;
  const int tid = threadIdx.x;
  const float* convw = d ? cwb : cwf;
  const float* convb = d ? cbb : cbf;
  const float* dtb   = d ? dtbb : dtbf;
  const float* Alog  = d ? Alb : Alf;
  const float* Dp    = d ? Dpb : Dpf;

  if (tid >= 384 && tid < 384 + NH) {
    int h = tid - 384;
    float raw = bf2f(zxb[(size_t)row * NPP + 640 + h]) + dtb[h];
    float dt  = (raw > 20.f) ? raw : __logf(1.f + __expf(raw));
    dt_s[h] = dt;
    dAG[((size_t)db * NH + h) * SEQ + l] = dt * (-__expf(Alog[h]));
  }
  __syncthreads();
  if (tid < 384) {
    int j = tid;
    float acc = convb[j];
#pragma unroll
    for (int k = 0; k < 4; ++k) {
      int ls = l - 3 + k;
      if (ls >= 0)
        acc += bf2f(zxb[((size_t)db * SEQ + ls) * NPP + DI + j]) * convw[j * 4 + k];
    }
    float v = acc / (1.f + __expf(-acc));  // SiLU
    if (j < DI) {
      int h = j / HD;
      xdt[(size_t)row * DI + j] = f2bf(v * dt_s[h]);
      Yf[(size_t)row * DI + j]  = Dp[h] * v;
    } else if (j < DI + DS) {
      Bb[(size_t)row * DS + (j - DI)] = f2bf(v);
    } else {
      Cb[(size_t)row * DS + (j - DI - DS)] = f2bf(v);
    }
  }
}

// ---------------- K3: intra-chunk SSD tile: Yd += (L.(C B^T)) Xdt ; st = Xdt^T (dec.B)
// block = 4 waves; wave w owns a 16-row strip of the 64x64 tile.
// The Xdt 64x64 tile is staged once into LDS (coalesced) and reused by both the
// Yd B-fragments and the transposed chunk-state A-fragments (ds reads instead of
// 256-element-stride scattered global u16 loads).
__global__ void kchunk(const bf16_t* __restrict__ xdt, const bf16_t* __restrict__ Bb,
                       const bf16_t* __restrict__ Cb, const float* __restrict__ dAG,
                       float* __restrict__ AcsG, float* __restrict__ cdG,
                       float* __restrict__ Yf, float* __restrict__ stG) {
  __shared__ float  Acs_s[CH];
  __shared__ bf16_t Mbf[CH * CH];
  __shared__ bf16_t decB[CH * CH];
  __shared__ bf16_t Xs[CH * HD];
  const int c = blockIdx.x, h = blockIdx.y, db = blockIdx.z;
  const int tid = threadIdx.x, lane = tid & 31, w = tid >> 5;
  const int l0 = c * CH, r0 = w * 16;
  const float* dA = dAG + ((size_t)db * NH + h) * SEQ + l0;

  if (tid == 0) {
    float s = 0.f;
    for (int i = 0; i < CH; ++i) { s += dA[i]; Acs_s[i] = s; }
    cdG[((size_t)db * NH + h) * NC + c] = __expf(s);
  }
  __syncthreads();
  if (tid < CH) AcsG[((size_t)db * NH + h) * SEQ + l0 + tid] = Acs_s[tid];

  const bf16_t* Bchunk = Bb + ((size_t)db * SEQ + l0) * DS;
  const bf16_t* Cchunk = Cb + ((size_t)db * SEQ + l0) * DS;
  const bf16_t* Xchunk = xdt + ((size_t)db * SEQ + l0) * DI + h * HD;
  const float last = Acs_s[CH - 1];
  for (int i = tid; i < CH * DS; i += 128) {
    int li = i / DS, n = i % DS;
    decB[i] = f2bf(__expf(last - Acs_s[li]) * bf2f(Bchunk[li * DS + n]));
  }
  // stage Xdt tile [l][p] into LDS (coalesced: consecutive tid -> consecutive p)
  for (int i = tid; i < CH * HD; i += 128) {
    int li = i / HD, p = i % HD;
    Xs[i] = Xchunk[(size_t)li * DI + p];
  }

  // M = L .* (C B^T)  -> Mbf (bf16 in LDS)
#pragma unroll
  for (int j = 0; j < 4; ++j) {
    v8f acc = {};
#pragma unroll
    for (int ks = 0; ks < DS; ks += 32) {
      v16bf a  = load_a_frag(Cchunk + r0 * DS + ks, DS, 1, lane);      // A[l][n]
      v16bf bb = load_b_frag(Bchunk + j * 16 * DS + ks, 1, DS, lane);  // B[n][s]=Bc[s][n]
      acc = wmma_bf16(a, bb, acc);
    }
    const int n = lane & 15, mb = (lane < 16) ? 0 : 8;
#pragma unroll
    for (int v = 0; v < 8; ++v) {
      int li = r0 + mb + v, si = j * 16 + n;
      float Lv = (si <= li) ? __expf(Acs_s[li] - Acs_s[si]) : 0.f;
      Mbf[li * CH + si] = f2bf(acc[v] * Lv);
    }
  }
  __syncthreads();

  // Yd strip: Y[l0+r0.., h*HD..] += M @ Xdt     (Xdt from LDS)
  float* Yc = Yf + ((size_t)db * SEQ + l0) * DI + h * HD;
#pragma unroll
  for (int j = 0; j < 4; ++j) {
    v8f acc = {};
#pragma unroll
    for (int ks = 0; ks < CH; ks += 32) {
      v16bf a  = load_a_frag(Mbf + r0 * CH + ks, CH, 1, lane);     // A[l][s]
      v16bf bb = load_b_frag(Xs + ks * HD + j * 16, HD, 1, lane);  // B[s][p]
      acc = wmma_bf16(a, bb, acc);
    }
    accum_c_f32(Yc + (size_t)r0 * DI + j * 16, DI, lane, acc);
  }

  // chunk state: st[p][n] = sum_l Xdt[l][p] * decB[l][n]   (Xdt^T from LDS)
  float* stC = stG + (((size_t)db * NC + c) * NH + h) * HD * DS;
#pragma unroll
  for (int j = 0; j < 4; ++j) {
    v8f acc = {};
#pragma unroll
    for (int ks = 0; ks < CH; ks += 32) {
      v16bf a  = load_a_frag(Xs + r0 + ks * HD, 1, HD, lane);       // A[p][l]=Xdt[l][p]
      v16bf bb = load_b_frag(decB + ks * DS + j * 16, DS, 1, lane); // B[l][n]
      acc = wmma_bf16(a, bb, acc);
    }
    store_c_f32(stC + r0 * DS + j * 16, DS, lane, acc);
  }
}

// ---------------- K4: inter-chunk scan (in place: stG[c] becomes prev-state) ----
__global__ void kscan(float* __restrict__ stG, const float* __restrict__ cdG) {
  const int bid = blockIdx.x;               // db*NH + h
  const int tid = threadIdx.x;
  const int db = bid / NH, h = bid % NH;
  float state[16];
#pragma unroll
  for (int i = 0; i < 16; ++i) state[i] = 0.f;
  for (int c = 0; c < NC; ++c) {
    float cd = cdG[(size_t)bid * NC + c];
    float* stc = stG + (((size_t)db * NC + c) * NH + h) * HD * DS;
#pragma unroll
    for (int i = 0; i < 16; ++i) {
      int idx = tid + i * 256;
      float v = stc[idx];
      stc[idx] = state[i];                  // prev-state out
      state[i] = state[i] * cd + v;
    }
  }
}

// ---------------- K5: Yo += exp(Acs[l]) * (C @ prev^T) ----------------
__global__ void kyo(const bf16_t* __restrict__ Cb, const float* __restrict__ stG,
                    const float* __restrict__ AcsG, float* __restrict__ Yf) {
  __shared__ bf16_t pbf[HD * DS];
  const int c = blockIdx.x, h = blockIdx.y, db = blockIdx.z;
  const int tid = threadIdx.x, lane = tid & 31, w = tid >> 5;
  const int l0 = c * CH, r0 = w * 16;
  const bf16_t* Cchunk = Cb + ((size_t)db * SEQ + l0) * DS;
  const float*  prev   = stG + (((size_t)db * NC + c) * NH + h) * HD * DS;
  const float*  Acs    = AcsG + ((size_t)db * NH + h) * SEQ + l0;
  float* Yc = Yf + ((size_t)db * SEQ + l0) * DI + h * HD;

  for (int i = tid; i < HD * DS; i += 128) pbf[i] = f2bf(prev[i]);
  __syncthreads();

#pragma unroll
  for (int j = 0; j < 4; ++j) {
    v8f acc = {};
#pragma unroll
    for (int ks = 0; ks < DS; ks += 32) {
      v16bf a  = load_a_frag(Cchunk + r0 * DS + ks, DS, 1, lane);     // A[l][n]
      v16bf bb = load_b_frag(pbf + j * 16 * DS + ks, 1, DS, lane);    // B[n][p]=prev[p][n]
      acc = wmma_bf16(a, bb, acc);
    }
    const int n = lane & 15, mb = (lane < 16) ? 0 : 8;
#pragma unroll
    for (int v = 0; v < 8; ++v) {
      int li = r0 + mb + v;
      Yc[(size_t)li * DI + j * 16 + n] += acc[v] * __expf(Acs[li]);
    }
  }
}

// ---------------- K6: y = Y*silu(z); RMS-norm * norm_w -> bf16 ----------------
__global__ void kgate(const float* __restrict__ Yf, const bf16_t* __restrict__ zxb,
                      const float* nwf, const float* nwb, bf16_t* __restrict__ ybf) {
  __shared__ float red[DI];
  const int row = blockIdx.x, j = threadIdx.x;
  const int d = row / (BATCH * SEQ);
  const float* nw = d ? nwb : nwf;
  float z = bf2f(zxb[(size_t)row * NPP + j]);
  float y = Yf[(size_t)row * DI + j] * (z / (1.f + __expf(-z)));
  red[j] = y * y;
  __syncthreads();
  for (int s = DI / 2; s > 0; s >>= 1) {
    if (j < s) red[j] += red[j + s];
    __syncthreads();
  }
  float scale = rsqrtf(red[0] / DI + 1e-5f);
  ybf[(size_t)row * DI + j] = f2bf(y * scale * nw[j]);
}

// ---------------- K7: out-proj GEMM; dir0 stores, dir1 accumulates flipped ----
__global__ void kgemm_out(const bf16_t* __restrict__ ybf, const bf16_t* __restrict__ wob,
                          float* __restrict__ out, int dir) {
  const int lane = threadIdx.x & 31;
  const int w    = threadIdx.x >> 5;
  const int row0 = blockIdx.x * 64 + w * 16;
  const int col0 = blockIdx.y * 64;
  const bf16_t* A  = ybf + ((size_t)dir * BATCH * SEQ + row0) * DI;
  const bf16_t* Wd = wob + (size_t)dir * DM * DI;
  v8f acc[4] = {};
  for (int ks = 0; ks < DI; ks += 32) {
    __builtin_prefetch(A + ks + 32, 0, 1);
    v16bf a = load_a_frag(A + ks, DI, 1, lane);
#pragma unroll
    for (int j = 0; j < 4; ++j) {
      v16bf bb = load_b_frag(Wd + (size_t)(col0 + 16 * j) * DI + ks, 1, DI, lane);
      acc[j] = wmma_bf16(a, bb, acc[j]);
    }
  }
  const int n = lane & 15, mb = (lane < 16) ? 0 : 8;
  const int b = row0 / SEQ;
#pragma unroll
  for (int j = 0; j < 4; ++j) {
#pragma unroll
    for (int v = 0; v < 8; ++v) {
      int l = (row0 + mb + v) % SEQ;
      int orow = b * SEQ + (dir ? (SEQ - 1 - l) : l);
      if (dir) out[(size_t)orow * DM + col0 + 16 * j + n] += acc[j][v];
      else     out[(size_t)orow * DM + col0 + 16 * j + n]  = acc[j][v];
    }
  }
}

// ---------------- host launcher ----------------
extern "C" void kernel_launch(void* const* d_in, const int* in_sizes, int n_in,
                              void* d_out, int out_size, void* d_ws, size_t ws_size,
                              hipStream_t stream) {
  const float* x    = (const float*)d_in[0];
  const float* Wi_f = (const float*)d_in[1];
  const float* cw_f = (const float*)d_in[2];
  const float* cb_f = (const float*)d_in[3];
  const float* db_f = (const float*)d_in[4];
  const float* Al_f = (const float*)d_in[5];
  const float* Dp_f = (const float*)d_in[6];
  const float* nw_f = (const float*)d_in[7];
  const float* Wo_f = (const float*)d_in[8];
  const float* Wi_b = (const float*)d_in[9];
  const float* cw_b = (const float*)d_in[10];
  const float* cb_b = (const float*)d_in[11];
  const float* db_b = (const float*)d_in[12];
  const float* Al_b = (const float*)d_in[13];
  const float* Dp_b = (const float*)d_in[14];
  const float* nw_b = (const float*)d_in[15];
  const float* Wo_b = (const float*)d_in[16];
  float* out = (float*)d_out;

  char* ws = (char*)d_ws;
  size_t off = 0;
  auto carve = [&](size_t bytes) -> void* {
    void* p = ws + off;
    off += (bytes + 255) & ~(size_t)255;
    return p;
  };
  const size_t ROWS = (size_t)2 * BATCH * SEQ;        // both directions
  bf16_t* xbf  = (bf16_t*)carve(ROWS * DM * 2);
  bf16_t* wib  = (bf16_t*)carve((size_t)2 * NPP * DM * 2);
  bf16_t* wob  = (bf16_t*)carve((size_t)2 * DM * DI * 2);
  bf16_t* zxb  = (bf16_t*)carve(ROWS * NPP * 2);
  bf16_t* xdt  = (bf16_t*)carve(ROWS * DI * 2);
  bf16_t* Bb   = (bf16_t*)carve(ROWS * DS * 2);
  bf16_t* Cb   = (bf16_t*)carve(ROWS * DS * 2);
  float*  dAG  = (float*)carve(ROWS * NH * 4);
  float*  AcsG = (float*)carve(ROWS * NH * 4);
  float*  cdG  = (float*)carve((size_t)2 * BATCH * NH * NC * 4);
  float*  Yf   = (float*)carve(ROWS * DI * 4);
  float*  stG  = (float*)carve((size_t)2 * BATCH * NC * NH * HD * DS * 4);
  bf16_t* ybf  = (bf16_t*)carve(ROWS * DI * 2);
  (void)ws_size; (void)in_sizes; (void)n_in; (void)out_size;

  kprep_x<<<(ROWS * DM + 255) / 256, 256, 0, stream>>>(x, xbf);
  kprep_wi<<<(2 * NPP * DM + 255) / 256, 256, 0, stream>>>(Wi_f, Wi_b, wib);
  kprep_wo<<<(2 * DM * DI + 255) / 256, 256, 0, stream>>>(Wo_f, Wo_b, wob);

  kgemm_in<<<dim3(ROWS / 64, NPP / 64), 128, 0, stream>>>(xbf, wib, zxb);

  kpost<<<ROWS, 512, 0, stream>>>(zxb, cw_f, cb_f, db_f, Al_f, Dp_f,
                                  cw_b, cb_b, db_b, Al_b, Dp_b,
                                  xdt, Bb, Cb, dAG, Yf);

  kchunk<<<dim3(NC, NH, 2 * BATCH), 128, 0, stream>>>(xdt, Bb, Cb, dAG, AcsG, cdG,
                                                      Yf, stG);

  kscan<<<2 * BATCH * NH, 256, 0, stream>>>(stG, cdG);

  kyo<<<dim3(NC, NH, 2 * BATCH), 128, 0, stream>>>(Cb, stG, AcsG, Yf);

  kgate<<<ROWS, DI, 0, stream>>>(Yf, zxb, nw_f, nw_b, ybf);

  kgemm_out<<<dim3(BATCH * SEQ / 64, DM / 64), 128, 0, stream>>>(ybf, wob, out, 0);
  kgemm_out<<<dim3(BATCH * SEQ / 64, DM / 64), 128, 0, stream>>>(ybf, wob, out, 1);
}
